// WM_LSTM_13194139533699
// MI455X (gfx1250) — matrix operand order repaired and checked by reading, not compiled, not run on hardware
//
#include <hip/hip_runtime.h>

// ---------------- problem constants ----------------
constexpr int kT   = 512;
constexpr int kB   = 64;
constexpr int kIn  = 256;
constexpr int kH   = 512;
constexpr int kG4  = 4 * kH;          // 2048
constexpr int kBlocks  = 16;          // 16 blocks x 8 waves = 128 waves = 128 C tiles
constexpr int kThreads = 256;

// ---------------- workspace layout (bytes) ----------------
constexpr size_t OFF_CNT = 0;
constexpr size_t OFF_H0  = 4096;
constexpr size_t OFF_H1  = OFF_H0  + (size_t)kB * kH * 2;
constexpr size_t OFF_XBF = OFF_H1  + (size_t)kB * kH * 2;
constexpr size_t OFF_WXT = OFF_XBF + (size_t)kT * kB * kIn * 2;
constexpr size_t OFF_WHT = OFF_WXT + (size_t)kIn * kG4 * 2;
// total ~19.2 MB

// LDS: per-block weight slice, resident for whole kernel
constexpr int    kLdsWhElems = 128 * kH;    // 128 rows of W_h^T (512 bf16 each) = 128 KB
constexpr int    kLdsWxElems = 128 * kIn;   // 128 rows of W_x^T (256 bf16 each) =  64 KB
constexpr size_t kLdsBytes   = (size_t)(kLdsWhElems + kLdsWxElems) * 2; // 192 KB (<320 KB/WGP)

typedef __attribute__((ext_vector_type(16))) __bf16 v16bf;
typedef __attribute__((ext_vector_type(8)))  __bf16 v8bf;
typedef __attribute__((ext_vector_type(8)))  float  v8f;

// ---- A fragment (16x32 bf16, row-major source, stride lda) ----
// ISA layout: lanes 0-15 row M=l hold K=0..7 (v0..3) and K=16..23 (v4..7);
//             lanes 16-31 row M=l hold K=8..15 and K=24..31.
__device__ __forceinline__ v16bf load_a_frag(const __bf16* __restrict__ p0,
                                             int lda, int l16, int hi) {
  const __bf16* p = p0 + l16 * lda + (hi ? 8 : 0);
  v8bf lo = *(const v8bf*)(p);
  v8bf up = *(const v8bf*)(p + 16);
  return __builtin_shufflevector(lo, up, 0,1,2,3,4,5,6,7,8,9,10,11,12,13,14,15);
}

// ---- B fragment (32x16 bf16) from LDS-resident W^T (row-major [N][K], stride ldk) ----
// ISA layout: lanes 0-15 col N=l hold K=0..7 and K=16..23; lanes 16-31 col N=l hold K=8..15 and K=24..31.
__device__ __forceinline__ v16bf load_b_lds(const __bf16* __restrict__ p0,
                                            int ldk, int l16, int hi) {
  const __bf16* p = p0 + l16 * ldk + (hi ? 8 : 0);
  v8bf lo = *(const v8bf*)(p);
  v8bf up = *(const v8bf*)(p + 16);
  return __builtin_shufflevector(lo, up, 0,1,2,3,4,5,6,7,8,9,10,11,12,13,14,15);
}

__device__ __forceinline__ float fast_sigmoid(float x) {
  return 1.0f / (1.0f + __expf(-x));
}
__device__ __forceinline__ float fast_tanh(float x) {
  x = fminf(fmaxf(x, -15.0f), 15.0f);
  float e = __expf(2.0f * x);
  return (e - 1.0f) / (e + 1.0f);
}

// ---- x_t @ W_x partial gates (independent of h -> overlapped with the barrier) ----
__device__ __forceinline__ void xpart_gates(const __bf16* __restrict__ xt,
                                            const __bf16* __restrict__ sWx,
                                            int jj0, int l16, int hi,
                                            v8f& a0, v8f& a1, v8f& a2, v8f& a3) {
  v8f z = {0,0,0,0,0,0,0,0};
  a0 = z; a1 = z; a2 = z; a3 = z;
  #pragma unroll
  for (int kk = 0; kk < kIn / 32; ++kk) {
    const int k0 = kk * 32;
    v16bf a   = load_a_frag(xt + k0, kIn, l16, hi);
    v16bf bb0 = load_b_lds(sWx + (0 * 32 + jj0) * kIn + k0, kIn, l16, hi);
    v16bf bb1 = load_b_lds(sWx + (1 * 32 + jj0) * kIn + k0, kIn, l16, hi);
    v16bf bb2 = load_b_lds(sWx + (2 * 32 + jj0) * kIn + k0, kIn, l16, hi);
    v16bf bb3 = load_b_lds(sWx + (3 * 32 + jj0) * kIn + k0, kIn, l16, hi);
    a0 = __builtin_amdgcn_wmma_f32_16x16x32_bf16(false, a, false, bb0, (short)0, a0, false, false);
    a1 = __builtin_amdgcn_wmma_f32_16x16x32_bf16(false, a, false, bb1, (short)0, a1, false, false);
    a2 = __builtin_amdgcn_wmma_f32_16x16x32_bf16(false, a, false, bb2, (short)0, a2, false, false);
    a3 = __builtin_amdgcn_wmma_f32_16x16x32_bf16(false, a, false, bb3, (short)0, a3, false, false);
  }
}

// ---------------- init: bf16 conversions, W transposes, h/barrier reset ----------------
__global__ void lstm_init(const float* __restrict__ x,
                          const float* __restrict__ Wx,
                          const float* __restrict__ Wh,
                          __bf16* __restrict__ x_bf,
                          __bf16* __restrict__ WxT,
                          __bf16* __restrict__ WhT,
                          __bf16* __restrict__ h0,
                          __bf16* __restrict__ h1,
                          unsigned* __restrict__ cnt) {
  const int stride = gridDim.x * blockDim.x;
  const int tid    = blockIdx.x * blockDim.x + threadIdx.x;
  for (int i = tid; i < kT * kB * kIn; i += stride) x_bf[i] = (__bf16)x[i];
  for (int i = tid; i < kIn * kG4; i += stride) {
    int k = i / kG4, n = i % kG4;                  // Wx[k][n]
    WxT[(size_t)n * kIn + k] = (__bf16)Wx[i];
  }
  for (int i = tid; i < kH * kG4; i += stride) {
    int k = i / kG4, n = i % kG4;                  // Wh[k][n]
    WhT[(size_t)n * kH + k] = (__bf16)Wh[i];
  }
  for (int i = tid; i < kB * kH; i += stride) { h0[i] = (__bf16)0.0f; h1[i] = (__bf16)0.0f; }
  if (tid == 0) *cnt = 0u;
}

// ---------------- persistent LSTM kernel ----------------
__global__ __launch_bounds__(kThreads, 1)
void lstm_persistent(const __bf16* __restrict__ x_bf,
                     const __bf16* __restrict__ WxT,
                     const __bf16* __restrict__ WhT,
                     const float*  __restrict__ bias,
                     __bf16* __restrict__ h0,
                     __bf16* __restrict__ h1,
                     float*  __restrict__ out,
                     unsigned* __restrict__ cnt) {
  extern __shared__ char smem_raw[];
  __bf16* sWh = (__bf16*)smem_raw;                             // [128][kH]
  __bf16* sWx = (__bf16*)(smem_raw + (size_t)kLdsWhElems * 2); // [128][kIn]

  const int tid  = threadIdx.x;
  const int lane = tid & 31;
  const int wave = tid >> 5;
  const int l16  = lane & 15;
  const int hi   = (lane >> 4) & 1;

  const int jbase = 32 * blockIdx.x;   // block owns hidden cols [jbase, jbase+32)
  const int m0    = (wave & 3) * 16;   // batch-row tile
  const int jj0   = (wave >> 2) * 16;  // local col tile (0 or 16)
  const int jcol  = jbase + jj0 + l16; // this lane's hidden column

  // ---- stage this block's weight slice into LDS (once, reused 512 steps) ----
  for (int idx = tid; idx < 128 * (kH / 8); idx += kThreads) {
    int r = idx / (kH / 8), c = idx % (kH / 8);
    int g = r >> 5, jj = r & 31;
    ((uint4*)sWh)[idx] =
        ((const uint4*)(WhT + (size_t)(g * kH + jbase + jj) * kH))[c];
  }
  for (int idx = tid; idx < 128 * (kIn / 8); idx += kThreads) {
    int r = idx / (kIn / 8), c = idx % (kIn / 8);
    int g = r >> 5, jj = r & 31;
    ((uint4*)sWx)[idx] =
        ((const uint4*)(WxT + (size_t)(g * kH + jbase + jj) * kIn))[c];
  }
  __syncthreads();

  // per-lane gate biases (loop-invariant)
  const float b_i = bias[0 * kH + jcol];
  const float b_f = bias[1 * kH + jcol];
  const float b_g = bias[2 * kH + jcol];
  const float b_o = bias[3 * kH + jcol];

  // cell state lives in registers for the entire sequence
  float creg[8];
  #pragma unroll
  for (int v = 0; v < 8; ++v) creg[v] = 0.0f;

  float* hsO = out;
  float* hTO = out + (size_t)kT * kB * kH;
  float* cTO = hTO + (size_t)kB * kH;

  // prologue: x-part gates for t = 0
  v8f xa0, xa1, xa2, xa3;
  xpart_gates(x_bf + (size_t)m0 * kIn, sWx, jj0, l16, hi, xa0, xa1, xa2, xa3);

  for (int t = 0; t < kT; ++t) {
    const __bf16* hRead  = (t & 1) ? h1 : h0;   // double-buffered hidden state
    __bf16*       hWrite = (t & 1) ? h0 : h1;

    // start from the pre-computed x-part
    v8f acc0 = xa0, acc1 = xa1, acc2 = xa2, acc3 = xa3;

    // ---- h_{t-1} @ W_h : K = 512 (the true serial dependency) ----
    const __bf16* hp = hRead + (size_t)m0 * kH;
    #pragma unroll 4
    for (int kk = 0; kk < kH / 32; ++kk) {
      const int k0 = kk * 32;
      v16bf a   = load_a_frag(hp + k0, kH, l16, hi);
      v16bf bb0 = load_b_lds(sWh + (0 * 32 + jj0) * kH + k0, kH, l16, hi);
      v16bf bb1 = load_b_lds(sWh + (1 * 32 + jj0) * kH + k0, kH, l16, hi);
      v16bf bb2 = load_b_lds(sWh + (2 * 32 + jj0) * kH + k0, kH, l16, hi);
      v16bf bb3 = load_b_lds(sWh + (3 * 32 + jj0) * kH + k0, kH, l16, hi);
      acc0 = __builtin_amdgcn_wmma_f32_16x16x32_bf16(false, a, false, bb0, (short)0, acc0, false, false);
      acc1 = __builtin_amdgcn_wmma_f32_16x16x32_bf16(false, a, false, bb1, (short)0, acc1, false, false);
      acc2 = __builtin_amdgcn_wmma_f32_16x16x32_bf16(false, a, false, bb2, (short)0, acc2, false, false);
      acc3 = __builtin_amdgcn_wmma_f32_16x16x32_bf16(false, a, false, bb3, (short)0, acc3, false, false);
    }

    // ---- fused LSTM elementwise (C-layout: element v -> row m0+v+hi*8, col jcol) ----
    #pragma unroll
    for (int v = 0; v < 8; ++v) {
      const int m = m0 + v + hi * 8;
      const float ig = fast_sigmoid(acc0[v] + b_i);
      const float fg = fast_sigmoid(acc1[v] + b_f);
      const float gg = fast_tanh   (acc2[v] + b_g);
      const float og = fast_sigmoid(acc3[v] + b_o);
      const float cn = fg * creg[v] + ig * gg;
      creg[v] = cn;
      const float hn = og * fast_tanh(cn);
      hWrite[(size_t)m * kH + jcol] = (__bf16)hn;
      hsO[(size_t)t * (kB * kH) + (size_t)m * kH + jcol] = hn;
      if (t == kT - 1) {
        hTO[(size_t)m * kH + jcol] = hn;
        cTO[(size_t)m * kH + jcol] = cn;
      }
    }

    // ---- split device-wide barrier; overlap next x-part with the wait ----
    if (t != kT - 1) {
      // arrive
      __threadfence();          // publish hWrite at agent scope
      __syncthreads();
      if (tid == 0) {
        __hip_atomic_fetch_add(cnt, 1u, __ATOMIC_RELEASE, __HIP_MEMORY_SCOPE_AGENT);
      }

      // overlapped work: x-part gates for t+1 (independent of h)
      xpart_gates(x_bf + (size_t)(t + 1) * (kB * kIn) + (size_t)m0 * kIn,
                  sWx, jj0, l16, hi, xa0, xa1, xa2, xa3);

      // wait
      if (tid == 0) {
        const unsigned target = (unsigned)(t + 1) * (unsigned)kBlocks;
        while (__hip_atomic_load(cnt, __ATOMIC_ACQUIRE, __HIP_MEMORY_SCOPE_AGENT) < target) {
          __builtin_amdgcn_s_sleep(2);
        }
      }
      __syncthreads();
    }
  }
}

extern "C" void kernel_launch(void* const* d_in, const int* in_sizes, int n_in,
                              void* d_out, int out_size, void* d_ws, size_t ws_size,
                              hipStream_t stream) {
  (void)in_sizes; (void)n_in; (void)out_size; (void)ws_size;
  const float* x  = (const float*)d_in[0];
  const float* Wx = (const float*)d_in[1];
  const float* Wh = (const float*)d_in[2];
  const float* b  = (const float*)d_in[3];
  float* out = (float*)d_out;
  char*  ws  = (char*)d_ws;

  unsigned* cnt = (unsigned*)(ws + OFF_CNT);
  __bf16* h0  = (__bf16*)(ws + OFF_H0);
  __bf16* h1  = (__bf16*)(ws + OFF_H1);
  __bf16* xbf = (__bf16*)(ws + OFF_XBF);
  __bf16* WxT = (__bf16*)(ws + OFF_WXT);
  __bf16* WhT = (__bf16*)(ws + OFF_WHT);

  (void)hipFuncSetAttribute((const void*)lstm_persistent,
                            hipFuncAttributeMaxDynamicSharedMemorySize,
                            (int)kLdsBytes);

  lstm_init<<<1024, 256, 0, stream>>>(x, Wx, Wh, xbf, WxT, WhT, h0, h1, cnt);
  lstm_persistent<<<kBlocks, kThreads, kLdsBytes, stream>>>(xbf, WxT, WhT, b, h0, h1, out, cnt);
}
